// FlowEmbedding_70325794505113
// MI455X (gfx1250) — matrix operand order repaired and checked by reading, not compiled, not run on hardware
//
#include <hip/hip_runtime.h>
#include <hip/hip_bf16.h>

// ---------------------------------------------------------------------------
// FlowEmbedding on MI455X (gfx1250, wave32):
//   K1: pack weights into per-lane WMMA B-fragment order (bf16)
//   K2: KNN top-16 (thread per query, LDS-tiled xyz2)
//   K3: wave-per-query fused MLP using v_wmma_f32_16x16x32_bf16 + maxpool
// ---------------------------------------------------------------------------

typedef __attribute__((ext_vector_type(16))) __bf16 v16bf;
typedef __attribute__((ext_vector_type(8)))  float  v8f;

union FragU { uint4 q[2]; v16bf v; };

#define BQ 8
#define NQ 4096
#define MQ 4096
#define CF 128
// layer-0 K padded 131 -> 160 (5 chunks of 32); layers 1,2: 128 (4 chunks)
#define NBLK 104              // (5+4+4) chunks * 8 ntiles

// --------------------------- weight packing --------------------------------
// Fragment block (layer l, chunk q, ntile t): lane ln, element e holds
// W[q*32 + (ln>>4)*8 + (e<8 ? e : e+8)][t*16 + (ln&15)]  (B-matrix 32x16 bf16 layout).
__global__ __launch_bounds__(256) void pack_weights_kernel(
    const float* __restrict__ W0, const float* __restrict__ W1,
    const float* __restrict__ W2, __bf16* __restrict__ wpack) {
  int tid = blockIdx.x * blockDim.x + threadIdx.x;
  if (tid >= NBLK * 32) return;
  int blk = tid >> 5, ln = tid & 31;
  int l, rel;
  if (blk < 40)      { l = 0; rel = blk; }
  else if (blk < 72) { l = 1; rel = blk - 40; }
  else               { l = 2; rel = blk - 72; }
  int q = rel >> 3, t = rel & 7;
  const float* W = (l == 0) ? W0 : ((l == 1) ? W1 : W2);
  int Kl = (l == 0) ? 131 : 128;
  int nn = t * 16 + (ln & 15);
  int kb = q * 32 + (ln >> 4) * 8;
  __bf16* dst = wpack + ((size_t)blk * 32 + ln) * 16;
#pragma unroll
  for (int e = 0; e < 16; e++) {
    int kk = kb + (e < 8 ? e : e + 8);
    float v = (kk < Kl) ? W[(size_t)kk * CF + nn] : 0.0f;
    dst[e] = (__bf16)v;
  }
}

// -------------------------------- KNN --------------------------------------
__global__ __launch_bounds__(256) void knn_kernel(
    const float* __restrict__ xyz1, const float* __restrict__ xyz2,
    int* __restrict__ idxOut) {
  __shared__ float sx[512], sy[512], sz[512];
  int b = blockIdx.x / (NQ / 256);
  int n = (blockIdx.x % (NQ / 256)) * 256 + threadIdx.x;
  const float* qp = xyz1 + ((size_t)b * NQ + n) * 3;
  float qx = qp[0], qy = qp[1], qz = qp[2];

  float bd[16]; int bi[16];
#pragma unroll
  for (int s = 0; s < 16; s++) { bd[s] = 3.4e38f; bi[s] = 0; }
  float worst = 3.4e38f; int ws = 0;

  for (int m0 = 0; m0 < MQ; m0 += 512) {
    for (int i = threadIdx.x; i < 512; i += 256) {
      const float* p = xyz2 + ((size_t)b * MQ + m0 + i) * 3;
      sx[i] = p[0]; sy[i] = p[1]; sz[i] = p[2];
    }
    __syncthreads();
#pragma unroll 4
    for (int j = 0; j < 512; j++) {
      float dx = sx[j] - qx, dy = sy[j] - qy, dz = sz[j] - qz;
      float d = fmaf(dx, dx, fmaf(dy, dy, dz * dz));
      if (d < worst) {
#pragma unroll
        for (int s = 0; s < 16; s++) if (s == ws) { bd[s] = d; bi[s] = m0 + j; }
        worst = bd[0]; ws = 0;
#pragma unroll
        for (int s = 1; s < 16; s++) if (bd[s] > worst) { worst = bd[s]; ws = s; }
      }
    }
    __syncthreads();
  }
  int* o = idxOut + ((size_t)b * NQ + n) * 16;
#pragma unroll
  for (int s = 0; s < 16; s++) o[s] = bi[s];
}

// ------------------------- fused MLP + maxpool -----------------------------
// One wave per query point. Activation tile 16x160 bf16 in wave-private LDS.
__global__ __launch_bounds__(256) void fe_mlp_kernel(
    const int* __restrict__ knn,
    const float* __restrict__ xyz1, const float* __restrict__ feat1,
    const float* __restrict__ xyz2, const float* __restrict__ feat2,
    const __bf16* __restrict__ wpack,
    const float* __restrict__ b0, const float* __restrict__ b1,
    const float* __restrict__ b2, float* __restrict__ outFeat) {
  __shared__ alignas(16) __bf16 sAct[8 * 16 * 160];
  const int ln = threadIdx.x & 31;
  const int w  = threadIdx.x >> 5;
  const int qid = blockIdx.x * 8 + w;          // b*NQ + n
  const int b = qid >> 12;                     // / 4096
  __bf16* act = sAct + w * (16 * 160);

  // ---- build layer-0 activation tile: [feat2[idx]*feat1 | xyz_diff | 0pad]
  float f1c[4];
  const float* f1p = feat1 + (size_t)qid * CF;
#pragma unroll
  for (int i = 0; i < 4; i++) f1c[i] = f1p[ln + i * 32];
  float qc = (ln < 3) ? xyz1[(size_t)qid * 3 + ln] : 0.0f;
  const int* nb = knn + (size_t)qid * 16;
  for (int j = 0; j < 16; j++) {
    int m = nb[j];
    const float* f2 = feat2 + ((size_t)b * MQ + m) * CF;
    __bf16* row = act + j * 160;
#pragma unroll
    for (int i = 0; i < 4; i++) row[ln + i * 32] = (__bf16)(f2[ln + i * 32] * f1c[i]);
    float xd = 0.0f;
    if (ln < 3) xd = xyz2[((size_t)b * MQ + m) * 3 + ln] - qc;
    row[128 + ln] = (__bf16)xd;                // cols 128..159 (131..159 = 0)
  }
  __syncthreads();

  const int r      = ln & 15;                  // A-matrix row / D column lane
  const int khalf  = (ln >> 4) * 8;            // A K-half offset; D row offset
  const int col0   = ln & 15;
  const v8f vzero = {0.f, 0.f, 0.f, 0.f, 0.f, 0.f, 0.f, 0.f};
  v8f acc[8];

  auto do_layer = [&](int chunks, int blkOff) {
#pragma unroll
    for (int t = 0; t < 8; t++) acc[t] = vzero;
    for (int q = 0; q < chunks; q++) {
      FragU a;
      const uint4* ap = (const uint4*)(act + r * 160 + q * 32 + khalf);
      a.q[0] = ap[0];          // K = khalf+0..7
      a.q[1] = ap[2];          // K = khalf+16..23   (+32 bytes)
#pragma unroll
      for (int t = 0; t < 8; t++) {
        FragU bf;
        const uint4* bp =
            (const uint4*)(wpack + ((size_t)(blkOff + q * 8 + t) * 32 + ln) * 16);
        bf.q[0] = bp[0]; bf.q[1] = bp[1];
        acc[t] = __builtin_amdgcn_wmma_f32_16x16x32_bf16(
            false, a.v, false, bf.v, (short)0, acc[t], false, false);
      }
    }
  };

  auto store_act = [&](const float* bias) {
#pragma unroll
    for (int t = 0; t < 8; t++) {
      float bb = bias[t * 16 + col0];
#pragma unroll
      for (int g = 0; g < 8; g++) {
        float v = fmaxf(acc[t][g] + bb, 0.0f);
        act[(g + khalf) * 160 + t * 16 + col0] = (__bf16)v;
      }
    }
  };

  do_layer(5, 0);   __syncthreads(); store_act(b0); __syncthreads();
  do_layer(4, 40);  __syncthreads(); store_act(b1); __syncthreads();
  do_layer(4, 72);

  // ---- bias + relu + max over the 16 neighbor rows, then store
#pragma unroll
  for (int t = 0; t < 8; t++) {
    float bb = b2[t * 16 + col0];
    float m = acc[t][0] + bb;
#pragma unroll
    for (int g = 1; g < 8; g++) m = fmaxf(m, acc[t][g] + bb);
    m = fmaxf(m, 0.0f);                        // relu(max) == max(relu)
    float other = __shfl_xor(m, 16, 32);       // combine M 0-7 with M 8-15
    m = fmaxf(m, other);
    if (ln < 16) outFeat[(size_t)qid * CF + t * 16 + ln] = m;
  }
}

// ------------------------------- launcher ----------------------------------
extern "C" void kernel_launch(void* const* d_in, const int* in_sizes, int n_in,
                              void* d_out, int out_size, void* d_ws, size_t ws_size,
                              hipStream_t stream) {
  (void)in_sizes; (void)n_in; (void)out_size; (void)ws_size;
  const float* xyz1  = (const float*)d_in[0];
  const float* feat1 = (const float*)d_in[1];
  const float* xyz2  = (const float*)d_in[2];
  const float* feat2 = (const float*)d_in[3];
  const float* W0 = (const float*)d_in[4]; const float* b0 = (const float*)d_in[5];
  const float* W1 = (const float*)d_in[6]; const float* b1 = (const float*)d_in[7];
  const float* W2 = (const float*)d_in[8]; const float* b2 = (const float*)d_in[9];
  float* out = (float*)d_out;

  int*    idxBuf = (int*)d_ws;
  __bf16* wpack  = (__bf16*)((char*)d_ws + (size_t)BQ * NQ * 16 * sizeof(int));

  // output tuple: (xyz1 [B,N,3], feat1_new [B,N,128]) flat-concatenated
  hipMemcpyAsync(out, xyz1, (size_t)BQ * NQ * 3 * sizeof(float),
                 hipMemcpyDeviceToDevice, stream);

  pack_weights_kernel<<<(NBLK * 32 + 255) / 256, 256, 0, stream>>>(W0, W1, W2, wpack);
  knn_kernel<<<BQ * (NQ / 256), 256, 0, stream>>>(xyz1, xyz2, idxBuf);
  fe_mlp_kernel<<<(BQ * NQ) / 8, 256, 0, stream>>>(
      idxBuf, xyz1, feat1, xyz2, feat2, wpack, b0, b1, b2,
      out + (size_t)BQ * NQ * 3);
}
